// ConvEP_24361054503172
// MI455X (gfx1250) — matrix operand order repaired and checked by reading, not compiled, not run on hardware
//
#include <hip/hip_runtime.h>

typedef __attribute__((ext_vector_type(16))) _Float16 v16h;
typedef __attribute__((ext_vector_type(8)))  _Float16 v8h;
typedef __attribute__((ext_vector_type(8)))  float    v8f;

union V16 { v16h v; v8h h[2]; };

#if defined(__gfx1250__) && __has_builtin(__builtin_amdgcn_global_load_async_to_lds_b128) && __has_builtin(__builtin_amdgcn_s_wait_asynccnt)
#define EP_ASYNC 1
typedef __attribute__((__vector_size__(16))) int ep_v4i;
#define EP_GPTR(p) ((__attribute__((address_space(1))) ep_v4i*)(p))
#define EP_LPTR(p) ((__attribute__((address_space(3))) ep_v4i*)(p))
#else
#define EP_ASYNC 0
#endif

__device__ __forceinline__ float rho_clip(float x) { return fminf(fmaxf(x, 0.f), 1.f); }

__device__ __forceinline__ v8f wmma_f16(const V16& a, const V16& b, v8f c) {
  return __builtin_amdgcn_wmma_f32_16x16x32_f16(false, a.v, false, b.v, (short)0, c, false, false);
}

// ---------------------------------------------------------------------------
// One-time kernels
// ---------------------------------------------------------------------------

// p1 = maxpool2x2(conv2d(data, Wc1) + bc1)   -> (256,64,12,12)
__global__ __launch_bounds__(256)
void ep_convbot(const float* __restrict__ data, const float* __restrict__ Wc1,
                const float* __restrict__ bc1, float* __restrict__ p1) {
  const int gid = blockIdx.x * 256 + threadIdx.x;            // < 2359296
  const int wp = gid % 12;
  const int hp = (gid / 12) % 12;
  const int c  = (gid / 144) % 64;
  const int b  = gid / 9216;
  const float* dptr = data + (size_t)b * 784;
  const float* wptr = Wc1 + c * 25;
  const float bb = bc1[c];
  float best = -3.4e38f;
#pragma unroll
  for (int dy = 0; dy < 2; ++dy)
#pragma unroll
    for (int dx = 0; dx < 2; ++dx) {
      const int h0 = 2 * hp + dy, w0 = 2 * wp + dx;
      float acc = bb;
#pragma unroll
      for (int kh = 0; kh < 5; ++kh)
#pragma unroll
        for (int kw = 0; kw < 5; ++kw)
          acc += dptr[(h0 + kh) * 28 + (w0 + kw)] * wptr[kh * 5 + kw];
      best = fmaxf(best, acc);
    }
  p1[gid] = best;
}

// Pack Wc0 (co,ci,5,5) into B-fragment layout for the FORWARD conv GEMM:
// out[((c*8+nt)*32+lane)*16+j], c = pos*2+kc2, K=ci = kc2*32+(lane<16?0:8)+(j&7)+(j&8?16:0),
// N=co = nt*16+(lane&15)
__global__ __launch_bounds__(256)
void ep_packW0(const float* __restrict__ Wc0, _Float16* __restrict__ out) {
  const int gid = blockIdx.x * 256 + threadIdx.x;            // < 204800
  const int j  = gid & 15;
  const int l  = (gid >> 4) & 31;
  const int nt = (gid >> 9) & 7;
  const int c  = gid >> 12;                                  // 0..49
  const int kc = (c & 1) << 5;
  const int pos = c >> 1;
  const int kh = pos / 5, kw = pos % 5;
  const int ci = kc + ((l < 16) ? 0 : 8) + (j & 7) + ((j & 8) ? 16 : 0);
  const int co = nt * 16 + (l & 15);
  out[gid] = (_Float16)Wc0[(((size_t)co * 64 + ci) * 5 + kh) * 5 + kw];
}

// Pack Wc0 into B-fragment layout for the TRANSPOSE conv GEMM:
// out[((c2*4+nt)*32+lane)*16+j], c2 = pos*4+kcb, K=co = kcb*32+(lane<16?0:8)+(j&7)+(j&8?16:0),
// N=ci = nt*16+(lane&15)
__global__ __launch_bounds__(256)
void ep_packW1(const float* __restrict__ Wc0, _Float16* __restrict__ out) {
  const int gid = blockIdx.x * 256 + threadIdx.x;            // < 204800
  const int j  = gid & 15;
  const int l  = (gid >> 4) & 31;
  const int nt = (gid >> 9) & 3;
  const int c2 = gid >> 11;                                  // 0..99
  const int kcb = c2 & 3;
  const int pos = c2 >> 2;
  const int kh = pos / 5, kw = pos % 5;
  const int co = kcb * 32 + ((l < 16) ? 0 : 8) + (j & 7) + ((j & 8) ? 16 : 0);
  const int ci = nt * 16 + (l & 15);
  out[gid] = (_Float16)Wc0[(((size_t)co * 64 + ci) * 5 + kh) * 5 + kw];
}

// ---------------------------------------------------------------------------
// Per-step kernels
// ---------------------------------------------------------------------------

// s2 (b,ci,12,12) f32 -> s2t (b,12,12,ci) f16  (channels-last for K-contiguity)
__global__ __launch_bounds__(256)
void ep_pack_s2(const float* __restrict__ s2, _Float16* __restrict__ s2t) {
  const int gid = blockIdx.x * 256 + threadIdx.x;            // < 2359296
  const int b = gid / 9216;
  const int rem = gid % 9216;
  const int ci = rem / 144;
  const int hw = rem % 144;
  s2t[((size_t)b * 144 + hw) * 64 + ci] = (_Float16)s2[gid];
}

// a0 = rho(s1 @ Wfc^T + bfc); s0n = 0.5 s0 + 0.5 a0;  td1 = s0 @ Wfc
__global__ __launch_bounds__(256)
void ep_fc(const float* __restrict__ s0cur, const float* __restrict__ s1cur,
           const float* __restrict__ Wfc, const float* __restrict__ bfc,
           float* __restrict__ s0next, float* __restrict__ td1) {
  __shared__ float sh_s1[2048];
  __shared__ float sh_s0[16];
  __shared__ float red[10 * 256];
  const int tid = threadIdx.x;
  const int b = blockIdx.x;
  for (int j = tid; j < 2048; j += 256) sh_s1[j] = s1cur[(size_t)b * 2048 + j];
  if (tid < 10) sh_s0[tid] = s0cur[b * 10 + tid];
  __syncthreads();
  float part[10];
#pragma unroll
  for (int o = 0; o < 10; ++o) part[o] = 0.f;
  for (int j = tid; j < 2048; j += 256) {
    const float v = sh_s1[j];
#pragma unroll
    for (int o = 0; o < 10; ++o) part[o] += v * Wfc[o * 2048 + j];
  }
#pragma unroll
  for (int o = 0; o < 10; ++o) red[o * 256 + tid] = part[o];
  __syncthreads();
  for (int s = 128; s > 0; s >>= 1) {
    if (tid < s) {
#pragma unroll
      for (int o = 0; o < 10; ++o) red[o * 256 + tid] += red[o * 256 + tid + s];
    }
    __syncthreads();
  }
  if (tid < 10) {
    const float a0 = rho_clip(red[tid * 256] + bfc[tid]);
    s0next[b * 10 + tid] = 0.5f * s0cur[b * 10 + tid] + 0.5f * a0;
  }
  for (int j = tid; j < 2048; j += 256) {
    float t = 0.f;
#pragma unroll
    for (int i = 0; i < 10; ++i) t += sh_s0[i] * Wfc[i * 2048 + j];
    td1[(size_t)b * 2048 + j] = t;
  }
}

// Forward conv as implicit GEMM via WMMA.
// M-tile = 64 rows (one batch image: m = b*64 + oh*8 + ow), N = co (8 waves x 16),
// K = 25 positions x 64ci in chunks of 32. Double-buffered LDS A-tile, async-to-LDS
// staging when available, 4 accumulators per wave (B reused 4x from registers).
__global__ __launch_bounds__(256)
void ep_convfwd(const _Float16* __restrict__ s2t, const _Float16* __restrict__ w0p,
                const float* __restrict__ bc0, float* __restrict__ y) {
  __shared__ __align__(16) unsigned short ldsA[2][64 * 32];
  const int tid  = threadIdx.x;
  const int lane = tid & 31;
  const int nt   = tid >> 5;            // 0..7
  const int b    = blockIdx.x;          // m0 = b*64

  const int lrow  = tid >> 2;           // loader row 0..63
  const int lpart = tid & 3;            // loader 16B segment
  const int loh = lrow >> 3, low = lrow & 7;

  const int r = lane & 15;
  const int koff = (lane < 16) ? 0 : 8;

  v8f acc[4];
#pragma unroll
  for (int s = 0; s < 4; ++s) acc[s] = (v8f){0.f,0.f,0.f,0.f,0.f,0.f,0.f,0.f};

  auto stage = [&](int c, int buf) {
    const int pos = c >> 1, kc = (c & 1) << 5;
    const int kh = pos / 5, kw = pos % 5;
    const _Float16* src =
        s2t + (((size_t)b * 12 + loh + kh) * 12 + (low + kw)) * 64 + kc + lpart * 8;
    unsigned short* dst = &ldsA[buf][lrow * 32 + lpart * 8];
#if EP_ASYNC
    __builtin_amdgcn_global_load_async_to_lds_b128(EP_GPTR(src), EP_LPTR(dst), 0, 0);
#else
    *(uint4*)dst = *(const uint4*)src;
#endif
  };

  stage(0, 0);
  for (int c = 0; c < 50; ++c) {
#if EP_ASYNC
    __builtin_amdgcn_s_wait_asynccnt(0);
#endif
    __syncthreads();
    const unsigned short* base = &ldsA[c & 1][0];
    V16 A[4];
#pragma unroll
    for (int s = 0; s < 4; ++s) {
      const unsigned short* lp = base + (s * 16 + r) * 32;
      A[s].h[0] = *(const v8h*)(lp + koff);
      A[s].h[1] = *(const v8h*)(lp + koff + 16);
    }
    if (c + 1 < 50) stage(c + 1, (c + 1) & 1);
    const _Float16* wp = w0p + ((size_t)(c * 8 + nt) * 32 + lane) * 16;
    V16 B;
    B.h[0] = *(const v8h*)(wp);
    B.h[1] = *(const v8h*)(wp + 8);
#pragma unroll
    for (int s = 0; s < 4; ++s) acc[s] = wmma_f16(A[s], B, acc[s]);
  }

  const int co = nt * 16 + r;
  const float bias = bc0[co];
  float* yb = y + (size_t)b * 8192 + co * 64;
#pragma unroll
  for (int s = 0; s < 4; ++s)
#pragma unroll
    for (int j = 0; j < 8; ++j) {
      const int row = s * 16 + ((lane < 16) ? j : (j + 8));
      yb[row] = acc[s][j] + bias;
    }
}

// maxpool + argmax over y; emit unpooled s1 scatter (f16, channels-last) and s1 update
__global__ __launch_bounds__(256)
void ep_pool(const float* __restrict__ y, const float* __restrict__ s1cur,
             const float* __restrict__ td1, _Float16* __restrict__ ut,
             float* __restrict__ s1next) {
  const int gid = blockIdx.x * 256 + threadIdx.x;            // < 524288
  const int owp = gid & 3;
  const int ohp = (gid >> 2) & 3;
  const int co  = (gid >> 4) & 127;
  const int b   = gid >> 11;
  const size_t ybase = (size_t)b * 8192 + co * 64 + (2 * ohp) * 8 + 2 * owp;
  const float v0 = y[ybase], v1 = y[ybase + 1], v2 = y[ybase + 8], v3 = y[ybase + 9];
  int sel = 0; float best = v0;
  if (v1 > best) { best = v1; sel = 1; }
  if (v2 > best) { best = v2; sel = 2; }
  if (v3 > best) { best = v3; sel = 3; }
  const float sv = s1cur[gid];
#pragma unroll
  for (int k = 0; k < 4; ++k) {
    const int dy = k >> 1, dx = k & 1;
    const size_t uidx = (((size_t)b * 8 + 2 * ohp + dy) * 8 + (2 * owp + dx)) * 128 + co;
    ut[uidx] = (k == sel) ? (_Float16)sv : (_Float16)0.f;
  }
  s1next[gid] = 0.5f * sv + 0.5f * rho_clip(best + td1[gid]);
}

// Transpose conv as implicit GEMM via WMMA, fused with the s2 state update.
// M-tile = 64 rows (m = b*144 + h*12 + w), N = ci (4 waves x 16),
// K = 25 positions x 128co in chunks of 32, boundary rows zero-filled in LDS.
__global__ __launch_bounds__(128)
void ep_convT(const _Float16* __restrict__ ut, const _Float16* __restrict__ w1p,
              const float* __restrict__ p1, const float* __restrict__ s2cur,
              float* __restrict__ s2next) {
  __shared__ __align__(16) unsigned short ldsA[2][64 * 32];
  const int tid  = threadIdx.x;
  const int lane = tid & 31;
  const int nt   = tid >> 5;            // 0..3
  const int m0   = blockIdx.x << 6;     // 64 rows

  const int lrow = tid >> 1;            // loader row 0..63
  const int lseg = (tid & 1) * 16;      // 32B segment (16 halves)
  const int lm = m0 + lrow;
  const int lb = lm / 144;
  const int lrem = lm % 144;
  const int lh = lrem / 12;
  const int lw = lrem % 12;

  const int r = lane & 15;
  const int koff = (lane < 16) ? 0 : 8;

  v8f acc[4];
#pragma unroll
  for (int s = 0; s < 4; ++s) acc[s] = (v8f){0.f,0.f,0.f,0.f,0.f,0.f,0.f,0.f};

  auto stage = [&](int c, int buf) {
    const int pos = c >> 2, kc = (c & 3) << 5;
    const int kh = pos / 5, kw = pos % 5;
    const int ih = lh - kh, iw = lw - kw;
    unsigned short* dst = &ldsA[buf][lrow * 32 + lseg];
    if (ih >= 0 && ih < 8 && iw >= 0 && iw < 8) {
      const _Float16* src = ut + (((size_t)lb * 8 + ih) * 8 + iw) * 128 + kc + lseg;
#if EP_ASYNC
      __builtin_amdgcn_global_load_async_to_lds_b128(EP_GPTR(src), EP_LPTR(dst), 0, 0);
      __builtin_amdgcn_global_load_async_to_lds_b128(EP_GPTR(src + 8), EP_LPTR(dst + 8), 0, 0);
#else
      *(uint4*)dst = *(const uint4*)src;
      *(uint4*)(dst + 8) = *(const uint4*)(src + 8);
#endif
    } else {
      const uint4 z = {0u, 0u, 0u, 0u};
      *(uint4*)dst = z;
      *(uint4*)(dst + 8) = z;
    }
  };

  stage(0, 0);
  for (int c = 0; c < 100; ++c) {
#if EP_ASYNC
    __builtin_amdgcn_s_wait_asynccnt(0);
#endif
    __syncthreads();
    const unsigned short* base = &ldsA[c & 1][0];
    V16 A[4];
#pragma unroll
    for (int s = 0; s < 4; ++s) {
      const unsigned short* lp = base + (s * 16 + r) * 32;
      A[s].h[0] = *(const v8h*)(lp + koff);
      A[s].h[1] = *(const v8h*)(lp + koff + 16);
    }
    if (c + 1 < 100) stage(c + 1, (c + 1) & 1);
    const _Float16* wp = w1p + ((size_t)(c * 4 + nt) * 32 + lane) * 16;
    V16 B;
    B.h[0] = *(const v8h*)(wp);
    B.h[1] = *(const v8h*)(wp + 8);
#pragma unroll
    for (int s = 0; s < 4; ++s) acc[s] = wmma_f16(A[s], B, acc[s]);
  }

  const int ci = nt * 16 + r;
#pragma unroll
  for (int s = 0; s < 4; ++s)
#pragma unroll
    for (int j = 0; j < 8; ++j) {
      const int row = s * 16 + ((lane < 16) ? j : (j + 8));
      const int m = m0 + row;
      const int b = m / 144;
      const int rem = m % 144;
      const size_t idx = ((size_t)b * 64 + ci) * 144 + rem;
      const float v = rho_clip(p1[idx] + acc[s][j]);
      s2next[idx] = 0.5f * s2cur[idx] + 0.5f * v;
    }
}

// ---------------------------------------------------------------------------
// Host driver
// ---------------------------------------------------------------------------
extern "C" void kernel_launch(void* const* d_in, const int* in_sizes, int n_in,
                              void* d_out, int out_size, void* d_ws, size_t ws_size,
                              hipStream_t stream) {
  (void)in_sizes; (void)n_in; (void)out_size; (void)ws_size;
  const float* data = (const float*)d_in[0];
  const float* s0i  = (const float*)d_in[1];
  const float* s1i  = (const float*)d_in[2];
  const float* s2i  = (const float*)d_in[3];
  const float* Wfc  = (const float*)d_in[4];
  const float* bfc  = (const float*)d_in[5];
  const float* Wc0  = (const float*)d_in[6];
  const float* bc0  = (const float*)d_in[7];
  const float* Wc1  = (const float*)d_in[8];
  const float* bc1  = (const float*)d_in[9];
  float* out = (float*)d_out;

  const size_t N_S0 = 2560, N_S1 = 524288, N_S2 = 2359296;
  char* ws = (char*)d_ws;
  size_t off = 0;
  auto alloc = [&](size_t bytes) -> char* {
    char* p = ws + off;
    off = (off + bytes + 255) & ~(size_t)255;
    return p;
  };
  float* s0A = (float*)alloc(N_S0 * 4);
  float* s0B = (float*)alloc(N_S0 * 4);
  float* s1A = (float*)alloc(N_S1 * 4);
  float* s1B = (float*)alloc(N_S1 * 4);
  float* s2A = (float*)alloc(N_S2 * 4);
  float* s2B = (float*)alloc(N_S2 * 4);
  float* p1  = (float*)alloc(N_S2 * 4);
  float* td1 = (float*)alloc(N_S1 * 4);
  float* ybuf = (float*)alloc((size_t)2097152 * 4);
  _Float16* s2t = (_Float16*)alloc((size_t)2359296 * 2);
  _Float16* ut  = (_Float16*)alloc((size_t)2097152 * 2);
  _Float16* w0p = (_Float16*)alloc((size_t)204800 * 2);
  _Float16* w1p = (_Float16*)alloc((size_t)204800 * 2);

  // One-time setup (re-done every call: deterministic)
  ep_packW0<<<800, 256, 0, stream>>>(Wc0, w0p);
  ep_packW1<<<800, 256, 0, stream>>>(Wc0, w1p);
  ep_convbot<<<9216, 256, 0, stream>>>(data, Wc1, bc1, p1);
  (void)hipMemcpyAsync(s0A, s0i, N_S0 * 4, hipMemcpyDeviceToDevice, stream);
  (void)hipMemcpyAsync(s1A, s1i, N_S1 * 4, hipMemcpyDeviceToDevice, stream);
  (void)hipMemcpyAsync(s2A, s2i, N_S2 * 4, hipMemcpyDeviceToDevice, stream);

  float *s0c = s0A, *s1c = s1A, *s2c = s2A;
  float *s0n = s0B, *s1n = s1B, *s2n = s2B;
  for (int t = 0; t < 10; ++t) {
    ep_pack_s2<<<9216, 256, 0, stream>>>(s2c, s2t);
    ep_fc<<<256, 256, 0, stream>>>(s0c, s1c, Wfc, bfc, s0n, td1);
    ep_convfwd<<<256, 256, 0, stream>>>(s2t, w0p, bc0, ybuf);
    ep_pool<<<2048, 256, 0, stream>>>(ybuf, s1c, td1, ut, s1n);
    ep_convT<<<576, 128, 0, stream>>>(ut, w1p, p1, s2c, s2n);
    float* tp;
    tp = s0c; s0c = s0n; s0n = tp;
    tp = s1c; s1c = s1n; s1n = tp;
    tp = s2c; s2c = s2n; s2n = tp;
  }

  (void)hipMemcpyAsync(out, s0c, N_S0 * 4, hipMemcpyDeviceToDevice, stream);
  (void)hipMemcpyAsync(out + N_S0, s1c, N_S1 * 4, hipMemcpyDeviceToDevice, stream);
  (void)hipMemcpyAsync(out + N_S0 + N_S1, s2c, N_S2 * 4, hipMemcpyDeviceToDevice, stream);
}
